// MultiHeadAttention_49770081026463
// MI455X (gfx1250) — matrix operand order repaired and checked by reading, not compiled, not run on hardware
//
#include <hip/hip_runtime.h>
#include <hip/hip_bf16.h>

// MHA: B=4, S=2048, D=2048, H=16, HD=128. fp32 in/out, bf16 WMMA internally.

#define Bc 4
#define Sc 2048
#define Dc 2048
#define Hc 16
#define HDc 128

typedef __attribute__((ext_vector_type(16))) __bf16 v16bf;
typedef __attribute__((ext_vector_type(8)))  __bf16 v8bf;
typedef __attribute__((ext_vector_type(8)))  float  v8f;

__device__ __forceinline__ unsigned short f2bf_bits(float f) {
    union { float f; unsigned u; } v; v.f = f;
    unsigned u = v.u;
    unsigned r = (u + 0x7FFFu + ((u >> 16) & 1u)) >> 16;   // round-to-nearest-even
    return (unsigned short)r;
}

__device__ __forceinline__ v8f vzero8() {
    v8f z;
#pragma unroll
    for (int i = 0; i < 8; ++i) z[i] = 0.0f;
    return z;
}

// ---------------------------------------------------------------------------
// CDNA5 async Global->LDS copy (ASYNCcnt path). Falls back to a VGPR round
// trip if the builtin is not declared by this toolchain.
// Builtin signature (from compiler diagnostic): param0 = int4 addrspace(1)*,
// so we cast to vector-of-4-int pointers in the right address spaces.
// ---------------------------------------------------------------------------
#if defined(__has_builtin)
#if __has_builtin(__builtin_amdgcn_global_load_async_to_lds_b128)
#define HAS_ASYNC_LDS 1
#endif
#endif
#ifndef HAS_ASYNC_LDS
#define HAS_ASYNC_LDS 0
#endif

__device__ __forceinline__ void copy16B_to_lds(const void* gptr, void* lptr) {
#if HAS_ASYNC_LDS
    typedef int v4i_t __attribute__((ext_vector_type(4)));
    typedef __attribute__((address_space(1))) v4i_t* gv4p;
    typedef __attribute__((address_space(3))) v4i_t* lv4p;
    __builtin_amdgcn_global_load_async_to_lds_b128((gv4p)gptr, (lv4p)lptr, 0, 0);
#else
    *(uint4*)lptr = *(const uint4*)gptr;
#endif
}

__device__ __forceinline__ void async_lds_wait() {
#if HAS_ASYNC_LDS
#if __has_builtin(__builtin_amdgcn_s_wait_asynccnt)
    __builtin_amdgcn_s_wait_asynccnt(0);
#else
    asm volatile("s_wait_asynccnt 0x0" ::: "memory");
#endif
#endif
}

// Build a 16x32 bf16 A-fragment for one lane.
// rowbase points at 32 contiguous K values (bf16) of this lane's row (r = lane&15).
// half = lane>>4. Contiguous layout: half0 holds K={0..15}, half1 holds K={16..31}.
__device__ __forceinline__ v16bf load_afrag(const unsigned short* rowbase, int half) {
    const v8bf l0 = *reinterpret_cast<const v8bf*>(rowbase + (half ? 16 : 0));
    const v8bf h0 = *reinterpret_cast<const v8bf*>(rowbase + (half ? 24 : 8));
    return __builtin_shufflevector(l0, h0, 0, 1, 2, 3, 4, 5, 6, 7, 8, 9, 10, 11, 12, 13, 14, 15);
}

// ---------------------------------------------------------------------------
// GEMM: Y[m][n] = sum_k A[m][k] * W[n][k]   (M=8192, N=2048, K=2048)
// A_BF16: A is bf16 (attn output) else fp32 (converted on LDS stage).
// OUT_MODE: 0 = bf16 [M][N]   (Q/K projections, head-major inside row)
//           1 = bf16 V-transposed: [b][h][hd][s]
//           2 = fp32 [M][N]   (final output projection)
// ---------------------------------------------------------------------------
template <bool A_BF16, int OUT_MODE>
__global__ __launch_bounds__(256) void gemm_wmma_kernel(const void* __restrict__ Aptr,
                                                        const float* __restrict__ W,
                                                        void* __restrict__ Out) {
    __shared__ __align__(64) unsigned short As[128 * 64];  // [row][k] bf16
    __shared__ __align__(64) unsigned short Bs[128 * 64];  // [n][k]  bf16

    const int tid  = threadIdx.x;
    const int wave = tid >> 5;
    const int lane = tid & 31;
    const int half = lane >> 4;
    const int r16  = lane & 15;
    const int wm   = wave >> 1;   // 0..3 -> 32 rows each
    const int wn   = wave & 1;    // 0..1 -> 64 cols each

    const int Mbase = blockIdx.y * 128;
    const int Nbase = blockIdx.x * 128;

    v8f acc[2][4];
#pragma unroll
    for (int mt = 0; mt < 2; ++mt)
#pragma unroll
        for (int nt = 0; nt < 4; ++nt) acc[mt][nt] = vzero8();

    const int row  = tid >> 1;
    const int part = (tid & 1) * 32;

    for (int k0 = 0; k0 < Dc; k0 += 64) {
        __syncthreads();
        // ---- stage A tile (128 x 64) ----
        if (A_BF16) {
            const unsigned short* src =
                (const unsigned short*)Aptr + (size_t)(Mbase + row) * Dc + k0 + part;
            unsigned short* dst = &As[row * 64 + part];
#pragma unroll
            for (int i = 0; i < 4; ++i) copy16B_to_lds(src + i * 8, dst + i * 8);
        } else {
            const float* src = (const float*)Aptr + (size_t)(Mbase + row) * Dc + k0 + part;
            unsigned* dst = (unsigned*)&As[row * 64 + part];
#pragma unroll
            for (int i = 0; i < 16; ++i) {
                unsigned lo = f2bf_bits(src[2 * i]);
                unsigned hi = f2bf_bits(src[2 * i + 1]);
                dst[i] = lo | (hi << 16);
            }
            __builtin_prefetch(src + 64, 0, 3);
        }
        // ---- stage B tile (128 x 64) from fp32 weights ----
        {
            const float* src = W + (size_t)(Nbase + row) * Dc + k0 + part;
            unsigned* dst = (unsigned*)&Bs[row * 64 + part];
#pragma unroll
            for (int i = 0; i < 16; ++i) {
                unsigned lo = f2bf_bits(src[2 * i]);
                unsigned hi = f2bf_bits(src[2 * i + 1]);
                dst[i] = lo | (hi << 16);
            }
            __builtin_prefetch(src + 64, 0, 3);
        }
        if (A_BF16) async_lds_wait();
        __syncthreads();

#pragma unroll
        for (int c = 0; c < 2; ++c) {
            v16bf af[2];
#pragma unroll
            for (int mt = 0; mt < 2; ++mt)
                af[mt] = load_afrag(&As[(wm * 32 + mt * 16 + r16) * 64 + c * 32], half);
            v16bf bfr[4];
#pragma unroll
            for (int nt = 0; nt < 4; ++nt)
                bfr[nt] = *(const v16bf*)&Bs[(wn * 64 + nt * 16 + r16) * 64 + c * 32 + half * 16];
#pragma unroll
            for (int mt = 0; mt < 2; ++mt)
#pragma unroll
                for (int nt = 0; nt < 4; ++nt)
                    acc[mt][nt] = __builtin_amdgcn_wmma_f32_16x16x32_bf16(
                        false, af[mt], false, bfr[nt], (short)0, acc[mt][nt], false, false);
        }
    }

    // ---- epilogue ----
#pragma unroll
    for (int mt = 0; mt < 2; ++mt)
#pragma unroll
        for (int nt = 0; nt < 4; ++nt)
#pragma unroll
            for (int j = 0; j < 8; ++j) {
                const int m = Mbase + wm * 32 + mt * 16 + j + 8 * half;
                const int n = Nbase + wn * 64 + nt * 16 + r16;
                const float val = acc[mt][nt][j];
                if (OUT_MODE == 0) {
                    ((unsigned short*)Out)[(size_t)m * Dc + n] = f2bf_bits(val);
                } else if (OUT_MODE == 1) {
                    const int b = m >> 11, s = m & 2047;
                    const int h = n >> 7, hd = n & 127;
                    ((unsigned short*)Out)[(((size_t)((b * Hc + h) * HDc + hd)) << 11) + s] =
                        f2bf_bits(val);
                } else {
                    ((float*)Out)[(size_t)m * Dc + n] = val;
                }
            }
}

// ---------------------------------------------------------------------------
// Flash attention.  Qh/Kh: bf16 [B][S][H*HD] (head-major within row);
// Vt: bf16 [B][H][HD][S].  Out: bf16 [B][S][H*HD].
// Block: one (b,h, 128 q rows); wave owns 16 q rows.
// ---------------------------------------------------------------------------
__global__ __launch_bounds__(256) void attn_wmma_kernel(const unsigned short* __restrict__ Qh,
                                                        const unsigned short* __restrict__ Kh,
                                                        const unsigned short* __restrict__ Vt,
                                                        const float* __restrict__ amask,
                                                        const float* __restrict__ kpad,
                                                        unsigned short* __restrict__ Ob) {
    __shared__ __align__(64) unsigned short Ks[32 * 128];   // [key][d]
    __shared__ __align__(64) unsigned short Vs[128 * 32];   // [d][key]
    __shared__ __align__(64) unsigned short Ps[8][16 * 32]; // per-wave P tile [row][key]

    const int tid  = threadIdx.x;
    const int wave = tid >> 5;
    const int lane = tid & 31;
    const int half = lane >> 4;
    const int r16  = lane & 15;

    const int qt = blockIdx.x & 15;
    const int bh = blockIdx.x >> 4;
    const int b  = bh >> 4;
    const int h  = bh & 15;
    const int qbase = qt * 128 + wave * 16;   // this wave's first q row

    // ---- Q fragments: 16 rows x 128 d, 4 K-chunks of 32 ----
    v16bf qf[4];
    {
        const unsigned short* qrow =
            Qh + (size_t)(b * Sc + qbase + r16) * Dc + h * HDc;
#pragma unroll
        for (int c = 0; c < 4; ++c) qf[c] = load_afrag(qrow + c * 32, half);
    }

    float mrun[8], lrun[8];
    v8f o[8];
#pragma unroll
    for (int j = 0; j < 8; ++j) { mrun[j] = -1e30f; lrun[j] = 0.0f; }
#pragma unroll
    for (int dt = 0; dt < 8; ++dt) o[dt] = vzero8();

    const float scale = 0.08838834764831845f;   // 1/sqrt(128)
    const int qa0 = qbase + 8 * half;           // absolute q of j==0 for this lane

    for (int kb = 0; kb < Sc; kb += 32) {
        __syncthreads();   // previous iteration done reading Ks/Vs
        {   // stage K tile: 32 keys x 128 d  (async Global->LDS, 2x b128 per thread)
            const int key = tid >> 3, kp = (tid & 7) * 16;
            const unsigned short* src = Kh + (size_t)(b * Sc + kb + key) * Dc + h * HDc + kp;
            unsigned short* dst = &Ks[key * 128 + kp];
            copy16B_to_lds(src, dst);
            copy16B_to_lds(src + 8, dst + 8);
            __builtin_prefetch((const void*)(src + 32 * Dc), 0, 3); // next key block
        }
        {   // stage Vt tile: 128 d x 32 keys
            const int d = tid >> 1, vp = (tid & 1) * 16;
            const unsigned short* src = Vt + (((size_t)(bh * HDc + d)) << 11) + kb + vp;
            unsigned short* dst = &Vs[d * 32 + vp];
            copy16B_to_lds(src, dst);
            copy16B_to_lds(src + 8, dst + 8);
            __builtin_prefetch((const void*)(src + 32), 0, 3);
        }
        async_lds_wait();
        __syncthreads();

        // ---- scores: two 16x16 tiles over this 32-key block ----
        v8f sc[2];
        sc[0] = vzero8(); sc[1] = vzero8();
#pragma unroll
        for (int t = 0; t < 2; ++t)
#pragma unroll
            for (int c = 0; c < 4; ++c) {
                v16bf bfr = *(const v16bf*)&Ks[(t * 16 + r16) * 128 + c * 32 + half * 16];
                sc[t] = __builtin_amdgcn_wmma_f32_16x16x32_bf16(
                    false, qf[c], false, bfr, (short)0, sc[t], false, false);
            }
        // ---- scale + masks ----
#pragma unroll
        for (int t = 0; t < 2; ++t) {
            const int kcol = kb + t * 16 + r16;
            const float kp_add = kpad[b * Sc + kcol];
#pragma unroll
            for (int j = 0; j < 8; ++j)
                sc[t][j] = sc[t][j] * scale + amask[(size_t)(qa0 + j) * Sc + kcol] + kp_add;
        }
        // ---- online softmax (stats per row, rows split across 16-lane halves) ----
        float alpha[8];
#pragma unroll
        for (int j = 0; j < 8; ++j) {
            float mx = fmaxf(sc[0][j], sc[1][j]);
            mx = fmaxf(mx, __shfl_xor(mx, 1));
            mx = fmaxf(mx, __shfl_xor(mx, 2));
            mx = fmaxf(mx, __shfl_xor(mx, 4));
            mx = fmaxf(mx, __shfl_xor(mx, 8));
            const float mnew = fmaxf(mrun[j], mx);
            alpha[j] = __expf(mrun[j] - mnew);
            mrun[j] = mnew;
            const float p0 = __expf(sc[0][j] - mnew);
            const float p1 = __expf(sc[1][j] - mnew);
            sc[0][j] = p0; sc[1][j] = p1;
            float rs = p0 + p1;
            rs += __shfl_xor(rs, 1);
            rs += __shfl_xor(rs, 2);
            rs += __shfl_xor(rs, 4);
            rs += __shfl_xor(rs, 8);
            lrun[j] = lrun[j] * alpha[j] + rs;
        }
#pragma unroll
        for (int dt = 0; dt < 8; ++dt)
#pragma unroll
            for (int j = 0; j < 8; ++j) o[dt][j] *= alpha[j];

        // ---- write P (bf16) to this wave's LDS tile, re-read in A layout ----
#pragma unroll
        for (int t = 0; t < 2; ++t)
#pragma unroll
            for (int j = 0; j < 8; ++j)
                Ps[wave][(j + 8 * half) * 32 + t * 16 + r16] = f2bf_bits(sc[t][j]);
        __syncthreads();
        const v16bf pf = load_afrag(&Ps[wave][r16 * 32], half);
#pragma unroll
        for (int dt = 0; dt < 8; ++dt) {
            v16bf vfr = *(const v16bf*)&Vs[(dt * 16 + r16) * 32 + half * 16];
            o[dt] = __builtin_amdgcn_wmma_f32_16x16x32_bf16(
                false, pf, false, vfr, (short)0, o[dt], false, false);
        }
    }

    // ---- finalize: divide by l, store bf16 [B][S][H*HD] ----
    float inv[8];
#pragma unroll
    for (int j = 0; j < 8; ++j) inv[j] = 1.0f / lrun[j];
#pragma unroll
    for (int dt = 0; dt < 8; ++dt)
#pragma unroll
        for (int j = 0; j < 8; ++j) {
            const int q = qa0 + j;
            const int e = h * HDc + dt * 16 + r16;
            Ob[(size_t)(b * Sc + q) * Dc + e] = f2bf_bits(o[dt][j] * inv[j]);
        }
}

// ---------------------------------------------------------------------------
extern "C" void kernel_launch(void* const* d_in, const int* in_sizes, int n_in,
                              void* d_out, int out_size, void* d_ws, size_t ws_size,
                              hipStream_t stream) {
    (void)in_sizes; (void)n_in; (void)out_size; (void)ws_size;
    const float* q     = (const float*)d_in[0];
    const float* k     = (const float*)d_in[1];
    const float* v     = (const float*)d_in[2];
    const float* amask = (const float*)d_in[3];
    const float* kpad  = (const float*)d_in[4];
    const float* wq    = (const float*)d_in[5];
    const float* wk    = (const float*)d_in[6];
    const float* wv    = (const float*)d_in[7];
    const float* wo    = (const float*)d_in[8];

    const size_t N = (size_t)Bc * Sc * Dc;          // 16,777,216 elements
    unsigned short* Qh = (unsigned short*)d_ws;     // bf16
    unsigned short* Kh = Qh + N;
    unsigned short* Vt = Kh + N;
    unsigned short* Ab = Vt + N;                    // attention output, bf16

    dim3 gg(Dc / 128, (Bc * Sc) / 128);             // (16, 64)
    dim3 blk(256);

    gemm_wmma_kernel<false, 0><<<gg, blk, 0, stream>>>((const void*)q, wq, (void*)Qh);
    gemm_wmma_kernel<false, 0><<<gg, blk, 0, stream>>>((const void*)k, wk, (void*)Kh);
    gemm_wmma_kernel<false, 1><<<gg, blk, 0, stream>>>((const void*)v, wv, (void*)Vt);

    attn_wmma_kernel<<<dim3(Bc * Hc * (Sc / 128)), blk, 0, stream>>>(Qh, Kh, Vt, amask, kpad, Ab);

    gemm_wmma_kernel<true, 2><<<gg, blk, 0, stream>>>((const void*)Ab, wo, d_out);
}
